// DSDAHead_46746424050089
// MI455X (gfx1250) — matrix operand order repaired and checked by reading, not compiled, not run on hardware
//
#include <hip/hip_runtime.h>

typedef __attribute__((ext_vector_type(16))) __bf16        v16bf;
typedef __attribute__((ext_vector_type(8)))  float         v8f;
typedef __attribute__((ext_vector_type(2)))  float         v2f;
typedef __attribute__((ext_vector_type(8)))  unsigned int  v8u;

#if defined(__has_builtin)
#if __has_builtin(__builtin_amdgcn_wmma_f32_16x16x4_f32)
#define HAVE_WMMA_F32K4 1
#endif
#endif

__device__ __forceinline__ unsigned short bfb(float f) {
    unsigned u = __float_as_uint(f);
    unsigned r = u + 0x7FFFu + ((u >> 16) & 1u);
    return (unsigned short)(r >> 16);
}
__device__ __forceinline__ __bf16 f2bf(float f) {
    unsigned short h = bfb(f);
    __bf16 o;
    __builtin_memcpy(&o, &h, 2);
    return o;
}
__device__ __forceinline__ unsigned int pk2(float a, float b) {
    return (unsigned int)bfb(a) | ((unsigned int)bfb(b) << 16);
}
__device__ __forceinline__ v8f v8f_zero() {
    v8f z = {0.f,0.f,0.f,0.f,0.f,0.f,0.f,0.f};
    return z;
}
// Build a 16-element bf16 fragment from two 16B chunks (dwords 0..3 = lo, 4..7 = hi)
__device__ __forceinline__ v16bf frag2(uint4 lo, uint4 hi) {
    v8u t;
    t[0] = lo.x; t[1] = lo.y; t[2] = lo.z; t[3] = lo.w;
    t[4] = hi.x; t[5] = hi.y; t[6] = hi.z; t[7] = hi.w;
    return __builtin_bit_cast(v16bf, t);
}
// q/k bf16 fragment (fallback path): 4 valid bf16 (K=0..3), upper half-wave zero
__device__ __forceinline__ v16bf qk_frag(uint2 raw, int hf) {
    v8u t = {0u,0u,0u,0u,0u,0u,0u,0u};
    t[0] = hf ? 0u : raw.x;
    t[1] = hf ? 0u : raw.y;
    return __builtin_bit_cast(v16bf, t);
}

#define WMMA_BF16(A,B,C) __builtin_amdgcn_wmma_f32_16x16x32_bf16(false,(A),false,(B),(short)0,(C),false,false)
#ifdef HAVE_WMMA_F32K4
#define WMMA_F32K4(A,B,C) __builtin_amdgcn_wmma_f32_16x16x4_f32(false,(A),false,(B),(short)0,(C),false,false)
#endif

// ---------------------------------------------------------------------------
// K1: f32 -> bf16 bulk convert (x -> xb)
// ---------------------------------------------------------------------------
__global__ __launch_bounds__(256) void cvt_bf16_kernel(const float* __restrict__ src,
                                                       __bf16* __restrict__ dst, int n) {
    int i = blockIdx.x * 256 + threadIdx.x;
    if (i < n) dst[i] = f2bf(src[i]);
}

// ---------------------------------------------------------------------------
// K0: weight prepack (bf16, [tap][M][C]) + BN folds
// BNF layout: [0..64) scale(5a|5c) [64..128) shift(5a|5c)
//             [128..160) s51 [160..192) sh51 [192..224) s52 [224..256) sh52
// ---------------------------------------------------------------------------
__global__ __launch_bounds__(256) void prepack_kernel(
    const float* __restrict__ w5a, const float* __restrict__ w5c,
    const float* __restrict__ w51, const float* __restrict__ w52,
    const float* s5a, const float* b5a, const float* m5a, const float* v5a,
    const float* s5c, const float* b5c, const float* m5c, const float* v5c,
    const float* s51, const float* b51, const float* m51, const float* v51,
    const float* s52, const float* b52, const float* m52, const float* v52,
    __bf16* __restrict__ WB2, __bf16* __restrict__ WB51, __bf16* __restrict__ WB52,
    float* __restrict__ BNF)
{
    int i = blockIdx.x * 256 + threadIdx.x;
    const int N2 = 9 * 64 * 512;      // 294912
    const int N51 = 9 * 32 * 32;      // 9216
    if (i < N2) {
        int t = i / (64 * 512); int rem = i - t * (64 * 512);
        int m = rem >> 9; int c = rem & 511;
        int ty = t / 3, tx = t % 3;
        float w = (m < 32) ? w5a[((m * 512 + c) * 3 + ty) * 3 + tx]
                           : w5c[(((m - 32) * 512 + c) * 3 + ty) * 3 + tx];
        WB2[i] = f2bf(w);
    } else if (i < N2 + N51) {
        int j = i - N2; int t = j / 1024; int m = (j >> 5) & 31; int c = j & 31;
        int ty = t / 3, tx = t % 3;
        WB51[j] = f2bf(w51[((m * 32 + c) * 3 + ty) * 3 + tx]);
    } else if (i < N2 + 2 * N51) {
        int j = i - N2 - N51; int t = j / 1024; int m = (j >> 5) & 31; int c = j & 31;
        int ty = t / 3, tx = t % 3;
        WB52[j] = f2bf(w52[((m * 32 + c) * 3 + ty) * 3 + tx]);
    } else if (i < N2 + 2 * N51 + 128) {
        int ch = i - N2 - 2 * N51;
        float s, bb, mm, vv; int si, shi;
        if (ch < 32)      { s = s5a[ch];     bb = b5a[ch];     mm = m5a[ch];     vv = v5a[ch];     si = ch;      shi = 64 + ch; }
        else if (ch < 64) { int c = ch - 32; s = s5c[c];       bb = b5c[c];      mm = m5c[c];      vv = v5c[c];  si = ch;      shi = 64 + ch; }
        else if (ch < 96) { int c = ch - 64; s = s51[c];       bb = b51[c];      mm = m51[c];      vv = v51[c];  si = 128 + c; shi = 160 + c; }
        else              { int c = ch - 96; s = s52[c];       bb = b52[c];      mm = m52[c];      vv = v52[c];  si = 192 + c; shi = 224 + c; }
        float inv = s * rsqrtf(vv + 1e-5f);
        BNF[si] = inv; BNF[shi] = bb - mm * inv;
    }
}

// ---------------------------------------------------------------------------
// K2/K5: implicit-GEMM 3x3 conv (pad 1) + BN + ReLU via WMMA bf16.
// ---------------------------------------------------------------------------
template<int MT, int CIN>
__global__ __launch_bounds__(256) void conv3x3_wmma(
    const __bf16* __restrict__ X, const __bf16* __restrict__ W,
    const float* __restrict__ bnf, int bn_off,
    float* __restrict__ out0, float* __restrict__ out1)
{
    constexpr int M = 16 * MT;
    constexpr int NW = 8 / MT;        // n-waves
    constexpr int NWID = 64 / NW;     // per-wave n width
    constexpr int JT = NWID / 16;     // WMMAs per K-step per wave

    const int bh = blockIdx.x; const int b = bh >> 6; const int h = bh & 63;
    const int tid = threadIdx.x; const int wid = tid >> 5; const int lane = tid & 31;
    const int mw = wid % MT; const int nw = wid / MT;
    const int hf = lane >> 4; const int l16 = lane & 15;

    __shared__ __bf16 lA[M][40];     // [m][k]
    __shared__ __bf16 lB2[64][40];   // [n][k]

    v8f acc[JT];
#pragma unroll
    for (int j = 0; j < JT; ++j) acc[j] = v8f_zero();

    for (int t = 0; t < 9; ++t) {
        int dy = t / 3 - 1, dx = t % 3 - 1;
        int hs = h + dy;
        bool rv = (unsigned)hs < 64u;
        const __bf16* Wt = W + t * (M * CIN);
        for (int c0 = 0; c0 < CIN; c0 += 32) {
            // stage A tile [M][32] as 16B chunks
            for (int i = tid; i < M * 4; i += 256) {
                int m = i >> 2, q = i & 3;
                uint4 v = *(const uint4*)(Wt + m * CIN + c0 + q * 8);
                *(uint4*)&lA[m][q * 8] = v;
            }
            // stage B tile transposed [n][k] with padding check
            {
                int k = tid >> 3; int nb = (tid & 7) * 8;
                int c = c0 + k;
                const __bf16* Xc = X + (((size_t)b * CIN + c) * 64 + (rv ? hs : 0)) * 64;
                if (c0 + 32 < CIN) __builtin_prefetch(Xc + 32 * 4096, 0, 0);
                __bf16 z = f2bf(0.f);
#pragma unroll
                for (int q = 0; q < 8; ++q) {
                    int n = nb + q; int w = n + dx;
                    lB2[n][k] = (rv && (unsigned)w < 64u) ? Xc[w] : z;
                }
            }
            __syncthreads();
            v16bf a;
            {
                int row = mw * 16 + l16;
                const uint4* pr = (const uint4*)&lA[row][0];
                a = frag2(pr[hf], pr[hf + 2]);
            }
#pragma unroll
            for (int j = 0; j < JT; ++j) {
                int col = nw * NWID + j * 16 + l16;
                const uint4* pb = (const uint4*)&lB2[col][0];
                v16bf bfrag = frag2(pb[hf], pb[hf + 2]);
                acc[j] = WMMA_BF16(a, bfrag, acc[j]);
            }
            __syncthreads();
        }
    }
    // epilogue: BN + ReLU, split channels
#pragma unroll
    for (int j = 0; j < JT; ++j) {
        int col = nw * NWID + j * 16 + l16;
#pragma unroll
        for (int r = 0; r < 8; ++r) {
            int m = mw * 16 + hf * 8 + r;
            float scale = bnf[bn_off + m];
            float shift = bnf[bn_off + M + m];
            float v = acc[j][r] * scale + shift;
            v = v > 0.f ? v : 0.f;
            float* o = (m < 32) ? out0 : out1;
            int mm = m & 31;
            o[(((size_t)b * 32 + mm) * 64 + h) * 64 + col] = v;
        }
    }
}

// ---------------------------------------------------------------------------
// K3: PAM q,k,v 1x1 convs.
// QP/KP: packed bf16 [B][4096] uint2 (fallback path).
// QP4/KP4: f32 [B][4096][4] (K=4 f32-WMMA path).  VT: bf16 c-major [B][32][4096]
// ---------------------------------------------------------------------------
__global__ __launch_bounds__(256) void qkv_kernel(
    const float* __restrict__ feat1,
    const float* qw, const float* qb, const float* kw, const float* kb,
    const float* vw, const float* vb,
    uint2* __restrict__ QP, uint2* __restrict__ KP,
    float* __restrict__ QP4, float* __restrict__ KP4,
    __bf16* __restrict__ vt)
{
    __shared__ float sqw[128], skw[128], svw[1024], sqb[4], skb[4], svb[32];
    int tid = threadIdx.x;
    for (int i = tid; i < 128; i += 256) { sqw[i] = qw[i]; skw[i] = kw[i]; }
    for (int i = tid; i < 1024; i += 256) svw[i] = vw[i];
    if (tid < 4) { sqb[tid] = qb[tid]; skb[tid] = kb[tid]; }
    if (tid < 32) svb[tid] = vb[tid];
    __syncthreads();

    int g = blockIdx.x * 256 + tid;       // 0..16383
    int b = g >> 12; int n = g & 4095;
    float f[32];
    const float* fp = feat1 + (size_t)b * 32 * 4096 + n;
#pragma unroll
    for (int c = 0; c < 32; ++c) f[c] = fp[c * 4096];
    float aq[4], ak[4];
#pragma unroll
    for (int o = 0; o < 4; ++o) {
        float q_ = sqb[o], k_ = skb[o];
#pragma unroll
        for (int c = 0; c < 32; ++c) { q_ += sqw[o * 32 + c] * f[c]; k_ += skw[o * 32 + c] * f[c]; }
        aq[o] = q_; ak[o] = k_;
    }
    QP[g] = make_uint2(pk2(aq[0], aq[1]), pk2(aq[2], aq[3]));
    KP[g] = make_uint2(pk2(ak[0], ak[1]), pk2(ak[2], ak[3]));
    *(float4*)(QP4 + (size_t)g * 4) = make_float4(aq[0], aq[1], aq[2], aq[3]);
    *(float4*)(KP4 + (size_t)g * 4) = make_float4(ak[0], ak[1], ak[2], ak[3]);
#pragma unroll
    for (int o = 0; o < 32; ++o) {
        float a = svb[o];
#pragma unroll
        for (int c = 0; c < 32; ++c) a += svw[o * 32 + c] * f[c];
        vt[((size_t)b * 32 + o) * 4096 + n] = f2bf(a);
    }
}

// ---------------------------------------------------------------------------
// K4: PAM flash attention. Block: 8 waves x 16 queries = 128 queries, all keys.
// Energy via V_WMMA_F32_16X16X4_F32 when available (K=4 exactly, f32 q/k),
// else zero-padded bf16 K=32 WMMA. PV via bf16 WMMA.
// ---------------------------------------------------------------------------
__global__ __launch_bounds__(256) void pam_kernel(
    const uint2* __restrict__ QP, const uint2* __restrict__ KP,
    const float* __restrict__ QP4, const float* __restrict__ KP4,
    const __bf16* __restrict__ vt, const float* __restrict__ feat1,
    const float* __restrict__ gamma_p, __bf16* __restrict__ sab)
{
    const int b = blockIdx.x >> 5;
    const int n0 = (blockIdx.x & 31) * 128;
    const int tid = threadIdx.x, wid = tid >> 5, lane = tid & 31;
    const int hf = lane >> 4, l16 = lane & 15;

    __shared__ __bf16 lVt[32][40];     // [c][m]
    __shared__ __bf16 lP[8][16][40];   // per-wave P tile [n][m]

#ifdef HAVE_WMMA_F32K4
    const float* qp4 = QP4 + (size_t)b * 4096 * 4;
    const float* kp4 = KP4 + (size_t)b * 4096 * 4;
    v2f aqf;
    {
        float2 t = *(const float2*)(qp4 + (size_t)(n0 + wid * 16 + l16) * 4 + hf * 2);
        aqf[0] = t.x; aqf[1] = t.y;
    }
#define LOAD_BK(dst, midx) { float2 t_ = *(const float2*)(kp4 + (size_t)(midx) * 4 + hf * 2); \
                             dst[0] = t_.x; dst[1] = t_.y; }
#define ENERGY(A,B) WMMA_F32K4((A),(B),v8f_zero())
    typedef v2f kfrag_t;
#else
    const uint2* qp = QP + (size_t)b * 4096;
    const uint2* kp = KP + (size_t)b * 4096;
    v16bf aqf = qk_frag(qp[n0 + wid * 16 + l16], hf);
#define LOAD_BK(dst, midx) dst = qk_frag(kp[(midx)], hf)
#define ENERGY(A,B) WMMA_BF16((A),(B),v8f_zero())
    typedef v16bf kfrag_t;
#endif

    float rmax[8];
#pragma unroll
    for (int r = 0; r < 8; ++r) rmax[r] = -3.0e38f;

    // ---- pass A: row maxima (no LDS, waves independent) ----
    for (int m0 = 0; m0 < 4096; m0 += 32) {
        kfrag_t bk0, bk1;
        LOAD_BK(bk0, m0 + l16);
        LOAD_BK(bk1, m0 + 16 + l16);
        v8f s0 = ENERGY(aqf, bk0);
        v8f s1 = ENERGY(aqf, bk1);
#pragma unroll
        for (int r = 0; r < 8; ++r) rmax[r] = fmaxf(rmax[r], fmaxf(s0[r], s1[r]));
    }
#pragma unroll
    for (int r = 0; r < 8; ++r)
        for (int msk = 1; msk < 16; msk <<= 1)
            rmax[r] = fmaxf(rmax[r], __shfl_xor(rmax[r], msk, 32));

    // ---- pass B: softmax numerators + O accumulation ----
    float rsum[8];
#pragma unroll
    for (int r = 0; r < 8; ++r) rsum[r] = 0.f;
    v8f o0 = v8f_zero(), o1 = v8f_zero();

    for (int m0 = 0; m0 < 4096; m0 += 32) {
        __syncthreads();
        {   // stage V block [c][m] : one uint2 (4 bf16) per thread
            int c = tid >> 3, mb = (tid & 7) * 4;
            uint2 v = *(const uint2*)(vt + ((size_t)b * 32 + c) * 4096 + m0 + mb);
            *(uint2*)&lVt[c][mb] = v;
        }
        __syncthreads();

        kfrag_t bk0, bk1;
        LOAD_BK(bk0, m0 + l16);
        LOAD_BK(bk1, m0 + 16 + l16);
        v8f s0 = ENERGY(aqf, bk0);
        v8f s1 = ENERGY(aqf, bk1);
#pragma unroll
        for (int r = 0; r < 8; ++r) {
            float p0 = __expf(s0[r] - rmax[r]);
            float p1 = __expf(s1[r] - rmax[r]);
            rsum[r] += p0 + p1;
            int row = hf * 8 + r;
            lP[wid][row][l16]      = f2bf(p0);
            lP[wid][row][16 + l16] = f2bf(p1);
        }
        // refragment P as A operand (same-wave LDS RAW; DS ops stay in order)
        v16bf ap;
        {
            const uint4* pp = (const uint4*)&lP[wid][l16][0];
            ap = frag2(pp[hf], pp[hf + 2]);
        }
        v16bf bv0, bv1;
        {
            const uint4* pv0 = (const uint4*)&lVt[l16][0];
            bv0 = frag2(pv0[hf], pv0[hf + 2]);
            const uint4* pv1 = (const uint4*)&lVt[16 + l16][0];
            bv1 = frag2(pv1[hf], pv1[hf + 2]);
        }
        o0 = WMMA_BF16(ap, bv0, o0);
        o1 = WMMA_BF16(ap, bv1, o1);
    }

#pragma unroll
    for (int r = 0; r < 8; ++r)
        for (int msk = 1; msk < 16; msk <<= 1)
            rsum[r] += __shfl_xor(rsum[r], msk, 32);

    float g = gamma_p[0];
#pragma unroll
    for (int j = 0; j < 2; ++j) {
        int c = j * 16 + l16;
#pragma unroll
        for (int r = 0; r < 8; ++r) {
            int n = n0 + wid * 16 + hf * 8 + r;
            float val = (j ? o1[r] : o0[r]) / rsum[r];
            size_t idx = ((size_t)b * 32 + c) * 4096 + n;
            sab[idx] = f2bf(g * val + feat1[idx]);
        }
    }
#undef LOAD_BK
#undef ENERGY
}

// ---------------------------------------------------------------------------
// K5a: CAM attention [32][32] per batch via WMMA (energy = F F^T), softmax
// ---------------------------------------------------------------------------
__global__ __launch_bounds__(256) void cam_att_kernel(const float* __restrict__ feat2,
                                                      float* __restrict__ att)
{
    const int b = blockIdx.x;
    const int tid = threadIdx.x, wid = tid >> 5, lane = tid & 31;
    const int hf = lane >> 4, l16 = lane & 15;
    __shared__ __bf16 lF[32][40];   // [c][n]
    __shared__ float  E[32][32];

    v8f acc = v8f_zero();
    const int mi = (wid >> 1) & 1, ni = wid & 1;

    for (int n0 = 0; n0 < 4096; n0 += 32) {
        __syncthreads();
        {   // stage: 4 f32 -> uint2 of bf16 per thread
            int c = tid >> 3, nb = (tid & 7) * 4;
            const float* sp = feat2 + ((size_t)b * 32 + c) * 4096 + n0 + nb;
            uint2 v = make_uint2(pk2(sp[0], sp[1]), pk2(sp[2], sp[3]));
            *(uint2*)&lF[c][nb] = v;
        }
        __syncthreads();
        if (wid < 4) {
            const uint4* pa = (const uint4*)&lF[mi * 16 + l16][0];
            v16bf a = frag2(pa[hf], pa[hf + 2]);
            const uint4* pb = (const uint4*)&lF[ni * 16 + l16][0];
            v16bf bb = frag2(pb[hf], pb[hf + 2]);
            acc = WMMA_BF16(a, bb, acc);
        }
    }
    if (wid < 4) {
#pragma unroll
        for (int r = 0; r < 8; ++r)
            E[mi * 16 + hf * 8 + r][ni * 16 + l16] = acc[r];
    }
    __syncthreads();
    if (tid < 32) {
        float mn = 3.0e38f;
        for (int d = 0; d < 32; ++d) mn = fminf(mn, E[tid][d]);
        float ss = 0.f;
        for (int d = 0; d < 32; ++d) ss += __expf(mn - E[tid][d]);
        float inv = 1.f / ss;
        for (int d = 0; d < 32; ++d)
            att[((size_t)b * 32 + tid) * 32 + d] = __expf(mn - E[tid][d]) * inv;
    }
}

// ---------------------------------------------------------------------------
// K5b: CAM out = att x F, residual, bf16 store
// ---------------------------------------------------------------------------
__global__ __launch_bounds__(256) void cam_out_kernel(const float* __restrict__ feat2,
                                                      const float* __restrict__ att,
                                                      const float* __restrict__ gamma_p,
                                                      __bf16* __restrict__ scb)
{
    __shared__ float sA[32][32];
    int b = blockIdx.x >> 4;
    int n = ((blockIdx.x & 15) * 256) + threadIdx.x;
    for (int i = threadIdx.x; i < 1024; i += 256) ((float*)sA)[i] = att[(size_t)b * 1024 + i];
    __syncthreads();
    float f[32];
    const float* fp = feat2 + (size_t)b * 32 * 4096 + n;
#pragma unroll
    for (int c = 0; c < 32; ++c) f[c] = fp[c * 4096];
    float g = gamma_p[0];
    __bf16* op = scb + (size_t)b * 32 * 4096 + n;
#pragma unroll
    for (int c = 0; c < 32; ++c) {
        float a = 0.f;
#pragma unroll
        for (int d = 0; d < 32; ++d) a += sA[c][d] * f[d];
        op[c * 4096] = f2bf(g * a + f[c]);
    }
}

// ---------------------------------------------------------------------------
// K6: feat_sum -> 1x1 conv (32->2) + bias + ReLU
// ---------------------------------------------------------------------------
__global__ __launch_bounds__(256) void final_kernel(const float* __restrict__ sa_conv,
                                                    const float* __restrict__ sc_conv,
                                                    const float* __restrict__ w8,
                                                    const float* __restrict__ b8,
                                                    float* __restrict__ out)
{
    __shared__ float sw[64], sb[2];
    int tid = threadIdx.x;
    if (tid < 64) sw[tid] = w8[tid];
    if (tid < 2) sb[tid] = b8[tid];
    __syncthreads();
    int g = blockIdx.x * 256 + tid; int b = g >> 12; int n = g & 4095;
    const float* pa = sa_conv + (size_t)b * 32 * 4096 + n;
    const float* pc = sc_conv + (size_t)b * 32 * 4096 + n;
    float a0 = sb[0], a1 = sb[1];
#pragma unroll
    for (int c = 0; c < 32; ++c) {
        float s = pa[c * 4096] + pc[c * 4096];
        a0 += sw[c] * s; a1 += sw[32 + c] * s;
    }
    out[((size_t)b * 2 + 0) * 4096 + n] = fmaxf(a0, 0.f);
    out[((size_t)b * 2 + 1) * 4096 + n] = fmaxf(a1, 0.f);
}

// ---------------------------------------------------------------------------
extern "C" void kernel_launch(void* const* d_in, const int* in_sizes, int n_in,
                              void* d_out, int out_size, void* d_ws, size_t ws_size,
                              hipStream_t stream) {
    const float* x     = (const float*)d_in[0];
    const float* w5a   = (const float*)d_in[1];
    const float* bn5a_s= (const float*)d_in[2];
    const float* bn5a_b= (const float*)d_in[3];
    const float* bn5a_m= (const float*)d_in[4];
    const float* bn5a_v= (const float*)d_in[5];
    const float* pq_w  = (const float*)d_in[6];
    const float* pq_b  = (const float*)d_in[7];
    const float* pk_w  = (const float*)d_in[8];
    const float* pk_b  = (const float*)d_in[9];
    const float* pv_w  = (const float*)d_in[10];
    const float* pv_b  = (const float*)d_in[11];
    const float* p_gamma=(const float*)d_in[12];
    const float* w51   = (const float*)d_in[13];
    const float* bn51_s= (const float*)d_in[14];
    const float* bn51_b= (const float*)d_in[15];
    const float* bn51_m= (const float*)d_in[16];
    const float* bn51_v= (const float*)d_in[17];
    const float* w5c   = (const float*)d_in[18];
    const float* bn5c_s= (const float*)d_in[19];
    const float* bn5c_b= (const float*)d_in[20];
    const float* bn5c_m= (const float*)d_in[21];
    const float* bn5c_v= (const float*)d_in[22];
    const float* c_gamma=(const float*)d_in[23];
    const float* w52   = (const float*)d_in[24];
    const float* bn52_s= (const float*)d_in[25];
    const float* bn52_b= (const float*)d_in[26];
    const float* bn52_m= (const float*)d_in[27];
    const float* bn52_v= (const float*)d_in[28];
    const float* w8    = (const float*)d_in[29];
    const float* b8    = (const float*)d_in[30];

    char* ws = (char*)d_ws;
    size_t off = 0;
    auto alloc = [&](size_t bytes) -> void* {
        void* p = ws + off;
        off = (off + bytes + 255) & ~(size_t)255;
        return p;
    };
    __bf16* XB    = (__bf16*)alloc(4ull * 512 * 4096 * 2);   // 16 MB
    __bf16* WB2   = (__bf16*)alloc(9ull * 64 * 512 * 2);
    __bf16* WB51  = (__bf16*)alloc(9ull * 32 * 32 * 2);
    __bf16* WB52  = (__bf16*)alloc(9ull * 32 * 32 * 2);
    float*  BNF   = (float*) alloc(256 * 4);
    float*  FEAT1 = (float*) alloc(4ull * 32 * 4096 * 4);
    float*  FEAT2 = (float*) alloc(4ull * 32 * 4096 * 4);
    uint2*  QP    = (uint2*) alloc(4ull * 4096 * 8);
    uint2*  KP    = (uint2*) alloc(4ull * 4096 * 8);
    float*  QP4   = (float*) alloc(4ull * 4096 * 4 * 4);
    float*  KP4   = (float*) alloc(4ull * 4096 * 4 * 4);
    __bf16* VT    = (__bf16*)alloc(4ull * 32 * 4096 * 2);
    __bf16* SAB   = (__bf16*)alloc(4ull * 32 * 4096 * 2);
    __bf16* SCB   = (__bf16*)alloc(4ull * 32 * 4096 * 2);
    float*  ATT   = (float*) alloc(4ull * 32 * 32 * 4);
    float*  SACONV= (float*) alloc(4ull * 32 * 4096 * 4);
    float*  SCCONV= (float*) alloc(4ull * 32 * 4096 * 4);

    cvt_bf16_kernel<<<32768, 256, 0, stream>>>(x, XB, 4 * 512 * 4096);
    prepack_kernel<<<1225, 256, 0, stream>>>(w5a, w5c, w51, w52,
        bn5a_s, bn5a_b, bn5a_m, bn5a_v,
        bn5c_s, bn5c_b, bn5c_m, bn5c_v,
        bn51_s, bn51_b, bn51_m, bn51_v,
        bn52_s, bn52_b, bn52_m, bn52_v,
        WB2, WB51, WB52, BNF);
    conv3x3_wmma<4, 512><<<256, 256, 0, stream>>>(XB, WB2, BNF, 0, FEAT1, FEAT2);
    qkv_kernel<<<64, 256, 0, stream>>>(FEAT1, pq_w, pq_b, pk_w, pk_b, pv_w, pv_b,
                                       QP, KP, QP4, KP4, VT);
    pam_kernel<<<128, 256, 0, stream>>>(QP, KP, QP4, KP4, VT, FEAT1, p_gamma, SAB);
    cam_att_kernel<<<4, 256, 0, stream>>>(FEAT2, ATT);
    cam_out_kernel<<<64, 256, 0, stream>>>(FEAT2, ATT, c_gamma, SCB);
    conv3x3_wmma<2, 32><<<256, 256, 0, stream>>>(SAB, WB51, BNF, 128, SACONV, SACONV);
    conv3x3_wmma<2, 32><<<256, 256, 0, stream>>>(SCB, WB52, BNF, 192, SCCONV, SCCONV);
    final_kernel<<<64, 256, 0, stream>>>(SACONV, SCCONV, w8, b8, (float*)d_out);
}
